// DiagonalizablePlant_27333171872193
// MI455X (gfx1250) — compile-verified, matrix operand order checked by reading, not run on hardware
//
#include <hip/hip_runtime.h>
#include <hip/hip_bf16.h>
#include <hip/hip_fp16.h>

// ---------------------------------------------------------------------------
// DiagonalizablePlant on MI455X (gfx1250, wave32, WMMA).
//   N_X=256, N_U=N_Y=32, B=16, T=8192.
// Compute-bound analysis: ~4.5 GFLOP over ~34 MB fused traffic would need
// ~3.1 PFLOP/s f32 to be BW-bound -> use f16 WMMA (16x16x32, f32 accum) for
// both GEMMs; keep the diagonal recurrence in f32. Chunked parallel scan
// (L=256, 32 chunks) exposes 512 independent workgroups.
// ---------------------------------------------------------------------------

typedef __attribute__((ext_vector_type(16))) _Float16 v16h;
typedef __attribute__((ext_vector_type(8)))  float    v8f;

#define NX    256
#define NU    32
#define NY    32
#define BATCH 16
#define TLEN  8192
#define LCH   256            // chunk length (= 2^8, used in lam^L)
#define NC    (TLEN / LCH)   // 32 chunks
#define NSUB  (LCH / 16)     // 16 subtiles of 16 time steps

// ---- WMMA fragment helpers (ISA 7.12.2 16-bit layouts) ---------------------
// 16-bit A (16xK=32): lane = M row (lane&15); half h -> K:
//   lanes 0-15 : h0..7 -> K0..7,  h8..15 -> K16..23
//   lanes 16-31: h0..7 -> K8..15, h8..15 -> K24..31
__device__ __forceinline__ int frag_k(int h, int lane) {
  return ((lane >> 4) << 3) + h + ((h >= 8) ? 8 : 0);
}

// A fragment from LDS row-major [16][stride] f16, K window at kOff.
__device__ __forceinline__ v16h load_a_lds(const _Float16* sm, int stride,
                                           int kOff, int lane) {
  int m = lane & 15;
  v16h a;
#pragma unroll
  for (int h = 0; h < 16; ++h) a[h] = sm[m * stride + kOff + frag_k(h, lane)];
  return a;
}

// B fragment (K=32 x N=16 tile) from global row-major [Krows][ncols] f16.
// lane = N column (lane&15); same per-lane K mapping as A.
__device__ __forceinline__ v16h load_b_glb(const _Float16* B, int ncols,
                                           int rowOff, int colBase, int lane) {
  int n = lane & 15;
  v16h b;
#pragma unroll
  for (int h = 0; h < 16; ++h)
    b[h] = B[(rowOff + frag_k(h, lane)) * ncols + colBase + n];
  return b;
}

// ---- K1: small projections  Bz^T, Cz^T, D^T (f16), z0 (f32) ----------------
__global__ __launch_bounds__(256)
void k1_prep(const float* __restrict__ x0, const float* __restrict__ Q,
             const float* __restrict__ Bm, const float* __restrict__ C,
             const float* __restrict__ D,
             _Float16* __restrict__ BzT, _Float16* __restrict__ CzT,
             _Float16* __restrict__ DT, float* __restrict__ z0) {
  int gid = blockIdx.x * blockDim.x + threadIdx.x;
  const int T0 = NU * NX;            // BzT  (32x256): BzT[j][n] = sum_k Q[k][n]*Bm[k][j]
  const int T1 = T0 + NX * NY;       // CzT  (256x32): CzT[n][y] = sum_k C[y][k]*Q[k][n]
  const int T2 = T1 + NU * NY;       // DT   (32x32):  DT[j][y]  = D[y][j]
  const int T3 = T2 + BATCH * NX;    // z0   (16x256): z0[b][n]  = sum_k x0[b][k]*Q[k][n]
  if (gid < T0) {
    int j = gid / NX, n = gid % NX;
    float acc = 0.f;
    for (int k = 0; k < NX; ++k) acc += Q[k * NX + n] * Bm[k * NU + j];
    BzT[j * NX + n] = (_Float16)acc;
  } else if (gid < T1) {
    int idx = gid - T0;
    int n = idx / NY, yy = idx % NY;
    float acc = 0.f;
    for (int k = 0; k < NX; ++k) acc += C[yy * NX + k] * Q[k * NX + n];
    CzT[n * NY + yy] = (_Float16)acc;
  } else if (gid < T2) {
    int idx = gid - T1;
    int j = idx / NY, yy = idx % NY;
    DT[j * NY + yy] = (_Float16)D[yy * NU + j];
  } else if (gid < T3) {
    int idx = gid - T2;
    int b = idx / NX, n = idx % NX;
    float acc = 0.f;
    for (int k = 0; k < NX; ++k) acc += x0[b * NX + k] * Q[k * NX + n];
    z0[b * NX + n] = acc;
  }
}

// ---- K2/K4: per-(batch,chunk) WMMA + scan pipeline -------------------------
// PASS 0: start z=0, accumulate chunk-local weighted sum w_c (chunk-end state).
// PASS 1: start z=Zstart[b][c], emit y for the chunk.
template <int PASS>
__global__ __launch_bounds__(256)
void k_chunk(const float* __restrict__ u, const float* __restrict__ lam,
             const _Float16* __restrict__ BzT, const _Float16* __restrict__ CzT,
             const _Float16* __restrict__ DT,
             const float* __restrict__ zstart,   // PASS1 in:  [B][NC][NX]
             float* __restrict__ wout,           // PASS0 out: [B][NC][NX]
             float* __restrict__ y) {            // PASS1 out: [B][T][NY]
  constexpr int US = NU + 8;   // 40  (f16 halfs)
  constexpr int VS = NX + 8;   // 264 (f32)
  constexpr int ZS = NX + 8;   // 264 (f16 halfs)
  __shared__ _Float16 sU[16 * US];
  __shared__ float    sV[16 * VS];
  __shared__ _Float16 sZ[16 * ZS];
  __shared__ float    sY[8 * 256];   // 8 waves x (16x16 f32) partials

  const int tid  = threadIdx.x;
  const int lane = tid & 31;
  const int w    = tid >> 5;          // wave 0..7
  const int b    = blockIdx.x / NC;
  const int c    = blockIdx.x % NC;
  const int t0   = c * LCH;

  // Invariant B fragments, held in registers for the whole chunk.
  // v-GEMM: wave w covers mode columns [32w, 32w+32).
  v16h bv0 = load_b_glb(BzT, NX, 0, (2 * w) * 16, lane);
  v16h bv1 = load_b_glb(BzT, NX, 0, (2 * w + 1) * 16, lane);
  const int ytile = w >> 2;           // output col tile 0..1 (16 cols each)
  const int ks    = w & 3;            // K slice: 64 modes each
  v16h by0 = {}, by1 = {}, bd = {};
  if (PASS == 1) {
    by0 = load_b_glb(CzT, NY, ks * 64,      ytile * 16, lane);
    by1 = load_b_glb(CzT, NY, ks * 64 + 32, ytile * 16, lane);
    if (ks == 0) bd = load_b_glb(DT, NY, 0, ytile * 16, lane);
  }

  const float lamn = lam[tid];        // thread tid owns mode n = tid
  float zn = (PASS == 1) ? zstart[((size_t)b * NC + c) * NX + tid] : 0.f;

  const float* uchunk = u + ((size_t)b * TLEN + t0) * NU;
  float* ychunk = (PASS == 1) ? (y + ((size_t)b * TLEN + t0) * NY) : nullptr;

  for (int s = 0; s < NSUB; ++s) {
    const float* ut = uchunk + s * 16 * NU;
    // Phase 1: stage u tile (16x32 f32 -> f16 LDS), coalesced 2 elems/thread.
    {
      int e0 = tid, e1 = tid + 256;
      float a0 = ut[e0], a1 = ut[e1];
      sU[(e0 >> 5) * US + (e0 & 31)] = (_Float16)a0;
      sU[(e1 >> 5) * US + (e1 & 31)] = (_Float16)a1;
      if (s + 1 < NSUB) __builtin_prefetch(ut + 16 * NU + 2 * tid, 0, 0);
    }
    __syncthreads();

    // Phase 2: v(16x256) = u(16x32) @ Bz^T(32x256); 2 WMMA tiles per wave.
    {
      v16h a = load_a_lds(sU, US, 0, lane);
      v8f c0 = {}, c1 = {};
      c0 = __builtin_amdgcn_wmma_f32_16x16x32_f16(false, a, false, bv0,
                                                  (short)0, c0, false, false);
      c1 = __builtin_amdgcn_wmma_f32_16x16x32_f16(false, a, false, bv1,
                                                  (short)0, c1, false, false);
      // C/D layout: N = lane&15, M = r + 8*(lane>=16).
      int n0 = (2 * w) * 16 + (lane & 15);
      int mb = (lane >> 4) * 8;
#pragma unroll
      for (int r = 0; r < 8; ++r) {
        sV[(mb + r) * VS + n0]      = c0[r];
        sV[(mb + r) * VS + n0 + 16] = c1[r];
      }
    }
    __syncthreads();

    // Phase 3: diagonal scan, 16 steps, f32. PASS1 records pre-update z_t.
#pragma unroll
    for (int j = 0; j < 16; ++j) {
      if (PASS == 1) sZ[j * ZS + tid] = (_Float16)zn;
      zn = __builtin_fmaf(lamn, zn, sV[j * VS + tid]);
    }

    if (PASS == 1) {
      __syncthreads();
      // Phase 4: Y(16x32) = Z(16x256) @ Cz^T + u @ D^T, split-K over 4 waves.
      v16h a0 = load_a_lds(sZ, ZS, ks * 64, lane);
      v16h a1 = load_a_lds(sZ, ZS, ks * 64 + 32, lane);
      v8f cy = {};
      cy = __builtin_amdgcn_wmma_f32_16x16x32_f16(false, a0, false, by0,
                                                  (short)0, cy, false, false);
      cy = __builtin_amdgcn_wmma_f32_16x16x32_f16(false, a1, false, by1,
                                                  (short)0, cy, false, false);
      if (ks == 0) {  // fold D-term into k-slice 0's partial
        v16h au = load_a_lds(sU, US, 0, lane);
        cy = __builtin_amdgcn_wmma_f32_16x16x32_f16(false, au, false, bd,
                                                    (short)0, cy, false, false);
      }
#pragma unroll
      for (int r = 0; r < 8; ++r) sY[w * 256 + lane * 8 + r] = cy[r];
      __syncthreads();

      // Phase 5: reduce 4 K-slices, coalesced store of 16x32 y tile.
      for (int e = tid; e < 512; e += 256) {
        int j = e >> 5, yy = e & 31;
        int tile = yy >> 4, n16 = yy & 15;
        int ln = n16 + ((j >= 8) ? 16 : 0);
        int r = j & 7;
        int base = tile * 4 * 256 + ln * 8 + r;
        float acc = sY[base] + sY[base + 256] + sY[base + 512] + sY[base + 768];
        ychunk[(s * 16 + j) * NY + yy] = acc;
      }
    }
    __syncthreads();
  }

  if (PASS == 0) wout[((size_t)b * NC + c) * NX + tid] = zn;
}

// ---- K3: cross-chunk scan  Z_{c+1} = lam^L * Z_c + w_c ---------------------
__global__ __launch_bounds__(256)
void k3_scan(const float* __restrict__ lam, const float* __restrict__ z0,
             const float* __restrict__ w, float* __restrict__ zstart) {
  int b = blockIdx.x;
  int n = threadIdx.x;
  float p = lam[n];
#pragma unroll
  for (int i = 0; i < 8; ++i) p = p * p;   // lam^256 (LCH = 2^8)
  float z = z0[b * NX + n];
  for (int c = 0; c < NC; ++c) {
    zstart[((size_t)b * NC + c) * NX + n] = z;
    z = __builtin_fmaf(p, z, w[((size_t)b * NC + c) * NX + n]);
  }
}

// ---------------------------------------------------------------------------
extern "C" void kernel_launch(void* const* d_in, const int* in_sizes, int n_in,
                              void* d_out, int out_size, void* d_ws,
                              size_t ws_size, hipStream_t stream) {
  (void)in_sizes; (void)n_in; (void)out_size; (void)ws_size;
  const float* x0  = (const float*)d_in[0];
  const float* u   = (const float*)d_in[1];
  const float* Q   = (const float*)d_in[2];
  const float* lam = (const float*)d_in[3];
  const float* Bm  = (const float*)d_in[4];
  const float* C   = (const float*)d_in[5];
  const float* D   = (const float*)d_in[6];
  float* y = (float*)d_out;

  // Workspace layout (~1.1 MB, all offsets 128B-aligned).
  char* p = (char*)d_ws;
  _Float16* BzT = (_Float16*)p; p += (size_t)NU * NX * sizeof(_Float16);   // 16 KB
  _Float16* CzT = (_Float16*)p; p += (size_t)NX * NY * sizeof(_Float16);   // 16 KB
  _Float16* DT  = (_Float16*)p; p += (size_t)NU * NY * sizeof(_Float16);   //  2 KB
  float* z0 = (float*)p; p += (size_t)BATCH * NX * sizeof(float);          // 16 KB
  float* w  = (float*)p; p += (size_t)BATCH * NC * NX * sizeof(float);     // 512 KB
  float* zs = (float*)p; p += (size_t)BATCH * NC * NX * sizeof(float);     // 512 KB

  const int prepTasks = NU * NX + NX * NY + NU * NY + BATCH * NX;  // 21504
  k1_prep<<<(prepTasks + 255) / 256, 256, 0, stream>>>(x0, Q, Bm, C, D,
                                                       BzT, CzT, DT, z0);
  k_chunk<0><<<BATCH * NC, 256, 0, stream>>>(u, lam, BzT, CzT, DT,
                                             nullptr, w, nullptr);
  k3_scan<<<BATCH, NX, 0, stream>>>(lam, z0, w, zs);
  k_chunk<1><<<BATCH * NC, 256, 0, stream>>>(u, lam, BzT, CzT, DT,
                                             zs, nullptr, y);
}
